// MultiHeadedAttention_79903571575325
// MI455X (gfx1250) — compile-verified, matrix operand order checked by reading, not run on hardware
//
#include <hip/hip_runtime.h>
#include <hip/hip_bf16.h>
#include <cstddef>

// ---------------------------------------------------------------------------
// MHA forward for MI455X (gfx1250, wave32, WMMA f32_16x16x32_f16)
// B=2, S=2048, D=1024, H=16, dh=64
// ---------------------------------------------------------------------------

#define BATCH 2
#define SEQ   2048
#define DMODEL 1024
#define NHEAD 16
#define DHEAD 64
#define TOK   (BATCH * SEQ)   // 4096

typedef __attribute__((ext_vector_type(16))) _Float16 v16h;
typedef __attribute__((ext_vector_type(8)))  _Float16 v8h;
typedef __attribute__((ext_vector_type(8)))  float    v8f;

union V16H { v16h v; v8h h[2]; };

__device__ __forceinline__ v8f zero8() {
  v8f z;
#pragma unroll
  for (int i = 0; i < 8; ++i) z[i] = 0.0f;
  return z;
}

// A/B fragment (16x32 f16, lane = M (or N for transposed-B), two 16B chunks).
// Per ISA: lanes 0-15 hold K {0..7,16..23}; lanes 16-31 hold K {8..15,24..31}.
__device__ __forceinline__ v16h load_frag_h16(const _Float16* __restrict__ base, int ld) {
  const int lane = threadIdx.x & 31;
  const _Float16* p = base + (size_t)(lane & 15) * ld + ((lane >> 4) << 3);
  V16H u;
  u.h[0] = *(const v8h*)(p);
  u.h[1] = *(const v8h*)(p + 16);
  return u.v;
}

// Same fragment but source is f32; convert inline (co-executes with WMMA).
__device__ __forceinline__ v16h load_frag_f32(const float* __restrict__ base, int ld) {
  const int lane = threadIdx.x & 31;
  const float* p = base + (size_t)(lane & 15) * ld + ((lane >> 4) << 3);
  v8f c0 = *(const v8f*)(p);
  v8f c1 = *(const v8f*)(p + 16);
  V16H u;
#pragma unroll
  for (int i = 0; i < 8; ++i) {
    u.h[0][i] = (_Float16)c0[i];
    u.h[1][i] = (_Float16)c1[i];
  }
  return u.v;
}

__device__ __forceinline__ v8f wmma16(v16h a, v16h b, v8f c) {
  return __builtin_amdgcn_wmma_f32_16x16x32_f16(false, a, false, b, (short)0, c,
                                                false, false);
}

// ---------------------------------------------------------------------------
// 1) f32 -> f16 convert (weights; stay hot in the 192MB L2)
// ---------------------------------------------------------------------------
__global__ __launch_bounds__(256) void cvt_kernel(const float* __restrict__ src,
                                                  _Float16* __restrict__ dst, int n) {
  int i = blockIdx.x * 256 + threadIdx.x;
  if (i < n) dst[i] = (_Float16)src[i];
}

// ---------------------------------------------------------------------------
// 2) Projection: Y = X @ W^T + b, write f16 head-major.
//    transposeOut==0 -> out[b,h,s,d]   (Q, K)
//    transposeOut==1 -> out[b,h,d,s]   (V, so PV B-loads are contiguous)
//    Block = 128x128 tile, 8 waves (4 in M x 2 in N), wave = 32x64.
// ---------------------------------------------------------------------------
__global__ __launch_bounds__(256) void proj_kernel(const float* __restrict__ X,
                                                   const _Float16* __restrict__ W,
                                                   const float* __restrict__ bias,
                                                   const unsigned char* __restrict__ mask,
                                                   _Float16* __restrict__ out,
                                                   int transposeOut) {
  const int w = threadIdx.x >> 5, lane = threadIdx.x & 31;
  const int waveM = blockIdx.y * 128 + (w & 3) * 32;
  const int waveN = blockIdx.x * 128 + (w >> 2) * 64;

  v8f acc[2][4];
#pragma unroll
  for (int mi = 0; mi < 2; ++mi)
#pragma unroll
    for (int ni = 0; ni < 4; ++ni) acc[mi][ni] = zero8();

  for (int k0 = 0; k0 < DMODEL; k0 += 32) {
    // prefetch the next K-step of the streaming activation rows
    __builtin_prefetch(X + (size_t)(waveM + (lane & 15)) * DMODEL + k0 + 64, 0, 3);
    v16h a[2], bf[4];
#pragma unroll
    for (int mi = 0; mi < 2; ++mi)
      a[mi] = load_frag_f32(X + (size_t)(waveM + mi * 16) * DMODEL + k0, DMODEL);
#pragma unroll
    for (int ni = 0; ni < 4; ++ni)
      bf[ni] = load_frag_h16(W + (size_t)(waveN + ni * 16) * DMODEL + k0, DMODEL);
#pragma unroll
    for (int mi = 0; mi < 2; ++mi)
#pragma unroll
      for (int ni = 0; ni < 4; ++ni) acc[mi][ni] = wmma16(a[mi], bf[ni], acc[mi][ni]);
  }

  const int rbase = (lane >> 4) * 8, cofs = lane & 15;
#pragma unroll
  for (int mi = 0; mi < 2; ++mi)
#pragma unroll
    for (int ni = 0; ni < 4; ++ni)
#pragma unroll
      for (int j = 0; j < 8; ++j) {
        int row = waveM + mi * 16 + rbase + j;   // token
        int col = waveN + ni * 16 + cofs;        // feature
        float v = acc[mi][ni][j] + bias[col];
        int b = row >> 11, s = row & (SEQ - 1);
        int h = col >> 6, d = col & (DHEAD - 1);
        if (mask[b * SEQ + s]) v = 0.0f;
        size_t idx = transposeOut
            ? ((size_t)((b * NHEAD + h) * DHEAD + d) * SEQ + s)
            : ((size_t)((b * NHEAD + h) * SEQ + s) * DHEAD + d);
        out[idx] = (_Float16)v;
      }
}

// ---------------------------------------------------------------------------
// 3) Fused scores + softmax: one block owns 16 full query rows.
//    attn[bh,q,k] = softmax_k( mask(QK^T * dh^-0.5) )   written ONCE.
//    8 waves x 16 N-tiles each -> 16 x 2048 logits in registers
//    (128 acc VGPRs/wave). Row max/sum: shfl butterfly within the 16-lane
//    half, then LDS reduction across the 8 waves. Causal tiles entirely
//    above the diagonal skip the WMMA (wave-uniform branch, EXEC stays full).
// ---------------------------------------------------------------------------
__global__ __launch_bounds__(256) void scores_softmax_kernel(
    const _Float16* __restrict__ Qh, const _Float16* __restrict__ Kh,
    float* __restrict__ attn, const unsigned char* __restrict__ kv_mask,
    const unsigned char* __restrict__ flag) {
  const int bh = blockIdx.y;
  const int m0 = blockIdx.x * 16;                 // first query row
  const int w = threadIdx.x >> 5, lane = threadIdx.x & 31;
  const int cbase = w * 256;                      // this wave's key-column slice
  const bool causal = flag[0] != 0;
  const int b = bh >> 4;

  const _Float16* q  = Qh + (size_t)bh * SEQ * DHEAD;
  const _Float16* kk = Kh + (size_t)bh * SEQ * DHEAD;
  float* obase = attn + (size_t)bh * SEQ * SEQ;

  // A fragments: 16 query rows, K = dh = 64
  v16h a0 = load_frag_h16(q + (size_t)m0 * DHEAD, DHEAD);
  v16h a1 = load_frag_h16(q + (size_t)m0 * DHEAD + 32, DHEAD);

  v8f acc[16];
#pragma unroll
  for (int nt = 0; nt < 16; ++nt) acc[nt] = zero8();

#pragma unroll
  for (int nt = 0; nt < 16; ++nt) {
    const int c0 = cbase + nt * 16;
    if (!(causal && c0 > m0 + 15)) {              // wave-uniform skip
      v16h b0 = load_frag_h16(kk + (size_t)c0 * DHEAD, DHEAD);
      v16h b1 = load_frag_h16(kk + (size_t)c0 * DHEAD + 32, DHEAD);
      acc[nt] = wmma16(a0, b0, acc[nt]);
      acc[nt] = wmma16(a1, b1, acc[nt]);
    }
  }

  // scale + mask -> logits in registers
  const float scale = 0.125f;                     // dh^-0.5
  const int rbase = (lane >> 4) * 8, cofs = lane & 15;
#pragma unroll
  for (int nt = 0; nt < 16; ++nt)
#pragma unroll
    for (int j = 0; j < 8; ++j) {
      int qi = m0 + rbase + j;
      int ki = cbase + nt * 16 + cofs;
      bool msk = causal ? (ki > qi) : (kv_mask[b * SEQ + ki] != 0);
      acc[nt][j] = msk ? -1.0e30f : acc[nt][j] * scale;
    }

  __shared__ float sred[8][16];

  // ---- row max ----
  float pm[8];
#pragma unroll
  for (int j = 0; j < 8; ++j) pm[j] = -3.4e38f;
#pragma unroll
  for (int nt = 0; nt < 16; ++nt)
#pragma unroll
    for (int j = 0; j < 8; ++j) pm[j] = fmaxf(pm[j], acc[nt][j]);
#pragma unroll
  for (int off = 8; off > 0; off >>= 1)
#pragma unroll
    for (int j = 0; j < 8; ++j) pm[j] = fmaxf(pm[j], __shfl_xor(pm[j], off, 32));
  if ((lane & 15) == 0) {
#pragma unroll
    for (int j = 0; j < 8; ++j) sred[w][rbase + j] = pm[j];
  }
  __syncthreads();
  float rowmax[8];
#pragma unroll
  for (int j = 0; j < 8; ++j) {
    float m = sred[0][rbase + j];
#pragma unroll
    for (int ww = 1; ww < 8; ++ww) m = fmaxf(m, sred[ww][rbase + j]);
    rowmax[j] = m;
  }
  __syncthreads();

  // ---- exp + row sum ----
  float ps[8];
#pragma unroll
  for (int j = 0; j < 8; ++j) ps[j] = 0.0f;
#pragma unroll
  for (int nt = 0; nt < 16; ++nt)
#pragma unroll
    for (int j = 0; j < 8; ++j) {
      float e = __expf(acc[nt][j] - rowmax[j]);
      acc[nt][j] = e;
      ps[j] += e;
    }
#pragma unroll
  for (int off = 8; off > 0; off >>= 1)
#pragma unroll
    for (int j = 0; j < 8; ++j) ps[j] += __shfl_xor(ps[j], off, 32);
  if ((lane & 15) == 0) {
#pragma unroll
    for (int j = 0; j < 8; ++j) sred[w][rbase + j] = ps[j];
  }
  __syncthreads();
  float inv[8];
#pragma unroll
  for (int j = 0; j < 8; ++j) {
    float s = sred[0][rbase + j];
#pragma unroll
    for (int ww = 1; ww < 8; ++ww) s += sred[ww][rbase + j];
    inv[j] = 1.0f / s;
  }

  // ---- single write of normalized weights ----
#pragma unroll
  for (int nt = 0; nt < 16; ++nt)
#pragma unroll
    for (int j = 0; j < 8; ++j) {
      int qi = m0 + rbase + j;
      int ki = cbase + nt * 16 + cofs;
      obase[(size_t)qi * SEQ + ki] = acc[nt][j] * inv[j];
    }
}

// ---------------------------------------------------------------------------
// 4) PV: x[bh,q,d] = sum_k P[bh,q,k] * V[bh,k,d]; V stored as [bh,d,k].
//    Block = 128 q-rows, 8 waves each own 16 rows x all 64 d-cols.
//    Causal: truncate the K loop past the diagonal (weights there are 0).
// ---------------------------------------------------------------------------
__global__ __launch_bounds__(256) void pv_kernel(const float* __restrict__ attn,
                                                 const _Float16* __restrict__ Vt,
                                                 _Float16* __restrict__ aout,
                                                 const unsigned char* __restrict__ flag) {
  const int bh = blockIdx.y;
  const int w = threadIdx.x >> 5, lane = threadIdx.x & 31;
  const int waveM = blockIdx.x * 128 + w * 16;
  const bool causal = flag[0] != 0;

  const float* P = attn + (size_t)bh * SEQ * SEQ + (size_t)waveM * SEQ;
  const _Float16* V = Vt + (size_t)bh * DHEAD * SEQ;

  v8f acc[4];
#pragma unroll
  for (int ni = 0; ni < 4; ++ni) acc[ni] = zero8();

  const int kend = causal ? ((waveM + 16 + 31) & ~31) : SEQ;
  for (int k0 = 0; k0 < kend; k0 += 32) {
    __builtin_prefetch(P + (size_t)(lane & 15) * SEQ + k0 + 64, 0, 3);
    v16h a = load_frag_f32(P + k0, SEQ);
    v16h bf[4];
#pragma unroll
    for (int ni = 0; ni < 4; ++ni)
      bf[ni] = load_frag_h16(V + (size_t)(ni * 16) * SEQ + k0, SEQ);
#pragma unroll
    for (int ni = 0; ni < 4; ++ni) acc[ni] = wmma16(a, bf[ni], acc[ni]);
  }

  const int b = bh >> 4, h = bh & 15;
  const int rbase = (lane >> 4) * 8, cofs = lane & 15;
#pragma unroll
  for (int ni = 0; ni < 4; ++ni)
#pragma unroll
    for (int j = 0; j < 8; ++j) {
      int s = waveM + rbase + j;
      int d = ni * 16 + cofs;
      aout[(size_t)(b * SEQ + s) * DMODEL + h * DHEAD + d] = (_Float16)acc[ni][j];
    }
}

// ---------------------------------------------------------------------------
// 5) Output projection: out = A @ Wo^T + bo, q_mask -> 0, f32 result.
// ---------------------------------------------------------------------------
__global__ __launch_bounds__(256) void outproj_kernel(const _Float16* __restrict__ A,
                                                      const _Float16* __restrict__ W,
                                                      const float* __restrict__ bias,
                                                      const unsigned char* __restrict__ qmask,
                                                      float* __restrict__ out) {
  const int w = threadIdx.x >> 5, lane = threadIdx.x & 31;
  const int waveM = blockIdx.y * 128 + (w & 3) * 32;
  const int waveN = blockIdx.x * 128 + (w >> 2) * 64;

  v8f acc[2][4];
#pragma unroll
  for (int mi = 0; mi < 2; ++mi)
#pragma unroll
    for (int ni = 0; ni < 4; ++ni) acc[mi][ni] = zero8();

  for (int k0 = 0; k0 < DMODEL; k0 += 32) {
    __builtin_prefetch(A + (size_t)(waveM + (lane & 15)) * DMODEL + k0 + 64, 0, 3);
    v16h a[2], bf[4];
#pragma unroll
    for (int mi = 0; mi < 2; ++mi)
      a[mi] = load_frag_h16(A + (size_t)(waveM + mi * 16) * DMODEL + k0, DMODEL);
#pragma unroll
    for (int ni = 0; ni < 4; ++ni)
      bf[ni] = load_frag_h16(W + (size_t)(waveN + ni * 16) * DMODEL + k0, DMODEL);
#pragma unroll
    for (int mi = 0; mi < 2; ++mi)
#pragma unroll
      for (int ni = 0; ni < 4; ++ni) acc[mi][ni] = wmma16(a[mi], bf[ni], acc[mi][ni]);
  }

  const int rbase = (lane >> 4) * 8, cofs = lane & 15;
#pragma unroll
  for (int mi = 0; mi < 2; ++mi)
#pragma unroll
    for (int ni = 0; ni < 4; ++ni)
#pragma unroll
      for (int j = 0; j < 8; ++j) {
        int row = waveM + mi * 16 + rbase + j;
        int col = waveN + ni * 16 + cofs;
        float v = acc[mi][ni][j] + bias[col];
        int b = row >> 11, s = row & (SEQ - 1);
        if (qmask[b * SEQ + s]) v = 0.0f;
        out[(size_t)row * DMODEL + col] = v;
      }
}

// ---------------------------------------------------------------------------
// Launch
// ---------------------------------------------------------------------------
extern "C" void kernel_launch(void* const* d_in, const int* in_sizes, int n_in,
                              void* d_out, int out_size, void* d_ws, size_t ws_size,
                              hipStream_t stream) {
  (void)in_sizes; (void)n_in; (void)out_size; (void)ws_size;

  const float* query = (const float*)d_in[0];
  const float* key_  = (const float*)d_in[1];
  const float* value = (const float*)d_in[2];
  const unsigned char* q_mask  = (const unsigned char*)d_in[3];
  const unsigned char* kv_mask = (const unsigned char*)d_in[4];
  const unsigned char* flag    = (const unsigned char*)d_in[5];
  const float* Wq = (const float*)d_in[6];
  const float* bq = (const float*)d_in[7];
  const float* Wk = (const float*)d_in[8];
  const float* bk = (const float*)d_in[9];
  const float* Wv = (const float*)d_in[10];
  const float* bv = (const float*)d_in[11];
  const float* Wo = (const float*)d_in[12];
  const float* bo = (const float*)d_in[13];

  float* out_x = (float*)d_out;                         // [B,S,D] = 4M floats
  float* attn  = out_x + (size_t)BATCH * SEQ * DMODEL;  // [B,H,S,S] = 128M floats

  const size_t WELEM = (size_t)DMODEL * DMODEL;  // 1M
  const size_t HELEM = (size_t)TOK * DMODEL;     // 4M
  _Float16* ws   = (_Float16*)d_ws;
  _Float16* Wq_h = ws;
  _Float16* Wk_h = Wq_h + WELEM;
  _Float16* Wv_h = Wk_h + WELEM;
  _Float16* Wo_h = Wv_h + WELEM;
  _Float16* Qh   = Wo_h + WELEM;   // [B,H,S,dh]
  _Float16* Kh   = Qh + HELEM;     // [B,H,S,dh]
  _Float16* Vt   = Kh + HELEM;     // [B,H,dh,S]
  _Float16* AO   = Vt + HELEM;     // [B,S,D] f16

  dim3 blk(256);

  // 1) weights -> f16
  int nW = (int)WELEM;
  cvt_kernel<<<(nW + 255) / 256, blk, 0, stream>>>(Wq, Wq_h, nW);
  cvt_kernel<<<(nW + 255) / 256, blk, 0, stream>>>(Wk, Wk_h, nW);
  cvt_kernel<<<(nW + 255) / 256, blk, 0, stream>>>(Wv, Wv_h, nW);
  cvt_kernel<<<(nW + 255) / 256, blk, 0, stream>>>(Wo, Wo_h, nW);

  // 2) projections (X converted f32->f16 inline in the fragment loads)
  dim3 gproj(DMODEL / 128, TOK / 128);  // (8, 32)
  proj_kernel<<<gproj, blk, 0, stream>>>(query, Wq_h, bq, q_mask, Qh, 0);
  proj_kernel<<<gproj, blk, 0, stream>>>(key_,  Wk_h, bk, kv_mask, Kh, 0);
  proj_kernel<<<gproj, blk, 0, stream>>>(value, Wv_h, bv, kv_mask, Vt, 1);

  // 3) fused masked scores + softmax (single pass over attn_weights)
  dim3 gss(SEQ / 16, BATCH * NHEAD);    // (128, 32)
  scores_softmax_kernel<<<gss, blk, 0, stream>>>(Qh, Kh, attn, kv_mask, flag);

  // 4) P @ V
  dim3 gpv(SEQ / 128, BATCH * NHEAD);   // (16, 32)
  pv_kernel<<<gpv, blk, 0, stream>>>(attn, Vt, AO, flag);

  // 5) output projection
  outproj_kernel<<<gproj, blk, 0, stream>>>(AO, Wo_h, bo, q_mask, out_x);
}
